// SelfAttention_4045859193701
// MI455X (gfx1250) — compile-verified, matrix operand order checked by reading, not run on hardware
//
#include <hip/hip_runtime.h>

// ---------------------------------------------------------------------------
// Fused self-attention for MI455X (gfx1250, wave32, WMMA 16x16x32 f16).
//  Pass 1: WMMA projections writing Q/K/V in *operand-ready swizzled* f16
//          layouts (each lane's 16 WMMA elements contiguous, 32B).
//  Pass 2: flash attention; K blocks via GLOBAL_LOAD_ASYNC_TO_LDS_B128
//          (ASYNCcnt), V blocks via TENSOR_LOAD_TO_LDS (TDM, TENSORcnt),
//          double-buffered in LDS.
// ---------------------------------------------------------------------------

typedef __attribute__((ext_vector_type(16))) _Float16     v16h;
typedef __attribute__((ext_vector_type(8)))  float        v8f;
typedef __attribute__((ext_vector_type(4)))  int          v4i;
typedef __attribute__((ext_vector_type(4)))  unsigned int u32x4;
typedef __attribute__((ext_vector_type(8)))  int          i32x8;
typedef __attribute__((ext_vector_type(4)))  int          i32x4;

#define HW  4096
#define CC  256
#define DQK 32

#if __has_builtin(__builtin_amdgcn_global_load_async_to_lds_b128) && \
    __has_builtin(__builtin_amdgcn_s_wait_asynccnt)
#define HAVE_ASYNC 1
#else
#define HAVE_ASYNC 0
#endif

#if __has_builtin(__builtin_amdgcn_tensor_load_to_lds) && \
    __has_builtin(__builtin_amdgcn_s_wait_tensorcnt)
#define HAVE_TDM 1
#else
#define HAVE_TDM 0
#endif

typedef __attribute__((address_space(1))) v4i gv4i;
typedef __attribute__((address_space(3))) v4i lv4i;

// 16-bit A/B operand lane/element -> K mapping (ISA 7.12.2):
// lanes 0-15 hold K {0..7,16..23}; lanes 16-31 hold K {8..15,24..31}.
__device__ __forceinline__ int kmap(int e, int grp) {
  int p = e >> 1, bit = e & 1;
  int base = (p < 4) ? ((grp ? 8 : 0) + 2 * p)
                     : ((grp ? 24 : 16) + 2 * (p - 4));
  return base + bit;
}
// Inverse: K value j -> (lane group, element)
__device__ __forceinline__ int ginv(int j) { return (j >> 3) & 1; }
__device__ __forceinline__ int einv(int j) { return (j & 7) + ((j >> 4) << 3); }

__device__ __forceinline__ v8f wmma_f16(v16h a, v16h b, v8f c) {
  return __builtin_amdgcn_wmma_f32_16x16x32_f16(
      false, a, false, b, (short)0, c, false, false);
}

// 16-byte LDS <- global copy via the async engine (ASYNCcnt).
__device__ __forceinline__ void cp16(void* lds, const void* gmem) {
#if HAVE_ASYNC
  __builtin_amdgcn_global_load_async_to_lds_b128(
      (gv4i*)(uintptr_t)gmem, (lv4i*)(uintptr_t)lds, 0, 0);
#else
  *(float4*)lds = *(const float4*)gmem;
#endif
}
__device__ __forceinline__ void wait_async() {
#if HAVE_ASYNC
  __builtin_amdgcn_s_wait_asynccnt(0);
#endif
}

#if HAVE_TDM
// 1-D TDM block copy global -> LDS: `bytes` must be a multiple of 8.
// D# per CDNA5 ISA ch.8: group0 {count=1, lds_addr, global_addr, type=2},
// group1 {data_size=3 (8B), tensor_dim0 >= tile, tile_dim0 = bytes/8, 1-D}.
__device__ __forceinline__ void tdm_load_1d(void* ldst, const void* gsrc,
                                            unsigned bytes) {
  const unsigned long long ga = (unsigned long long)(uintptr_t)gsrc;
  const unsigned lds = (unsigned)(uintptr_t)ldst;   // low 32 bits = LDS offset
  const unsigned n8 = bytes >> 3;                   // 8-byte elements
  u32x4 g0;
  g0[0] = 1u;                                       // count=1, user descriptor
  g0[1] = lds;                                      // lds_addr
  g0[2] = (unsigned)(ga & 0xffffffffu);             // global_addr[31:0]
  g0[3] = (unsigned)((ga >> 32) & 0x01ffffffu) | (2u << 30); // addr[56:32]|type=2
  i32x8 g1;
  g1[0] = (int)(3u << 16);                          // mask=0, data_size=8B
  g1[1] = (int)((n8 & 0xffffu) << 16);              // tensor_dim0[15:0]
  g1[2] = (int)((n8 >> 16) | (1u << 16));           // tensor_dim0[31:16]|dim1=1
  g1[3] = (int)((n8 & 0xffffu) << 16);              // tile_dim0 = n8
  g1[4] = 0;                                        // tile_dim1=0 (1-D)
  g1[5] = (int)n8;                                  // tensor_dim0_stride
  g1[6] = 0;
  g1[7] = 0;
  const i32x4 z4 = {0, 0, 0, 0};
#if defined(__clang_major__) && __clang_major__ >= 23
  const i32x8 z8 = {0, 0, 0, 0, 0, 0, 0, 0};
  __builtin_amdgcn_tensor_load_to_lds(g0, g1, z4, z4, z8, 0);
#else
  __builtin_amdgcn_tensor_load_to_lds(g0, g1, z4, z4, 0);
#endif
}
__device__ __forceinline__ void wait_tensor() {
  __builtin_amdgcn_s_wait_tensorcnt(0);
}
#endif

// ---------------------------------------------------------------------------
// Projection: out = swizzle(w @ x + bias) as f16.
// mode 0 (Q/K, Nout=32): per 16-pixel tile, A/B-operand layout
// mode 1 (V, Nout=256): per 32-key block x 16-ch tile, B-operand layout
// ---------------------------------------------------------------------------
__global__ __launch_bounds__(32) void proj_f16_kernel(
    const float* __restrict__ x, const float* __restrict__ w,
    const float* __restrict__ bias, _Float16* __restrict__ out,
    int Cin, int mode)
{
  const int b   = blockIdx.z;
  const int n0  = blockIdx.x * 16;   // pixel tile
  const int o0  = blockIdx.y * 16;   // output-channel tile
  const int lane = threadIdx.x & 31;
  const int col = lane & 15, grp = lane >> 4;
  const float* xb = x + (size_t)b * Cin * HW;

  v8f acc = {};
  for (int kc = 0; kc < Cin; kc += 32) {
    v16h a, bm;
#pragma unroll
    for (int e = 0; e < 16; ++e) {
      const int kd = kmap(e, grp);
      a[e]  = (_Float16)xb[(size_t)(kc + kd) * HW + (n0 + col)]; // A[m=pixel,k=ch]
      bm[e] = (_Float16)w[(size_t)(o0 + col) * Cin + (kc + kd)]; // B[k=ch,n=out]
    }
    acc = wmma_f16(a, bm, acc);
  }
  const int d  = o0 + col;           // output channel of this lane
  const float bi = bias[d];

  if (mode == 0) {                   // Q/K operand swizzle (d is the K-dim)
    _Float16* base = out + ((size_t)b * (HW / 16) + (n0 >> 4)) * 512;
    const int gd = ginv(d), ed = einv(d);
#pragma unroll
    for (int i = 0; i < 8; ++i) {
      const int m = i + grp * 8;     // pixel row within tile
      base[(gd * 16 + m) * 16 + ed] = (_Float16)(acc[i] + bi);
    }
  } else {                           // V operand swizzle (key is the K-dim)
    const int thi = d >> 4, cl = d & 15;
#pragma unroll
    for (int i = 0; i < 8; ++i) {
      const int m = i + grp * 8;
      const int n = n0 + m;          // key (pixel) index
      const int j = n & 31;
      out[(((size_t)b * (HW / 32) + (n >> 5)) * 16 + thi) * 512
          + (ginv(j) * 16 + cl) * 16 + einv(j)] = (_Float16)(acc[i] + bi);
    }
  }
}

// ---------------------------------------------------------------------------
// Flash attention. 4 waves/WG, 16 query rows per wave. 32-key blocks,
// double-buffered: K staged by async-to-LDS, V staged by the TDM.
// ---------------------------------------------------------------------------
__global__ __launch_bounds__(128) void attn_kernel(
    const _Float16* __restrict__ Q, const _Float16* __restrict__ K,
    const _Float16* __restrict__ V, const float* __restrict__ inp,
    const float* __restrict__ gamma, float* __restrict__ out)
{
  __shared__ __align__(32) _Float16 sK[2][1024];   // 2 x (32 keys x 32 dims)
  __shared__ __align__(32) _Float16 sV[2][8192];   // 2 x (32 keys x 256 ch)
  __shared__ __align__(32) _Float16 sP[4][512];    // per-wave P (A-layout)

  const int b    = blockIdx.y;
  const int tid  = threadIdx.x;
  const int wave = tid >> 5;
  const int lane = tid & 31;
  const int col  = lane & 15, grp = lane >> 4;
  const int i0   = blockIdx.x * 64 + wave * 16;    // query tile base

  const _Float16* Kb = K + (size_t)b * (HW / 16) * 512;   // swizzled K
  const _Float16* Vb = V + (size_t)b * (HW / 32) * 8192;  // swizzled V

  // Q operand: lane's 16 elements are contiguous 32B in the swizzled layout.
  const v16h qa = *(const v16h*)(Q + ((size_t)b * (HW / 16) + (i0 >> 4)) * 512
                                 + (size_t)lane * 16);

  v8f oacc[16];
  const v8f vzero = {};
#pragma unroll
  for (int t = 0; t < 16; ++t) oacc[t] = vzero;
  float rmax[8], rsum[8];
#pragma unroll
  for (int i = 0; i < 8; ++i) { rmax[i] = -1e30f; rsum[i] = 0.0f; }

  const int gc = ginv(col);          // P-store swizzle constants
  const int eb = col & 7;
  _Float16* myP = &sP[wave][0];

  // ---- stage block 0 ----
  cp16(&sK[0][tid * 8], Kb + tid * 8);
#if HAVE_TDM
  if (wave == 0) tdm_load_1d(&sV[0][0], Vb, 8192 * 2);
#else
#pragma unroll
  for (int r = 0; r < 8; ++r)
    cp16(&sV[0][(r * 128 + tid) * 8], Vb + (r * 128 + tid) * 8);
#endif

  int buf = 0;
  for (int j0 = 0; j0 < HW; j0 += 32, buf ^= 1) {
#if HAVE_TDM
    if (wave == 0) wait_tensor();
#endif
    wait_async();
    __syncthreads();                 // buf ready; buf^1 free for refill

    if (j0 + 32 < HW) {              // stage next block into the other buffer
      const _Float16* kblk = Kb + ((size_t)(j0 + 32) >> 4) * 512;
      const _Float16* vblk = Vb + ((size_t)(j0 + 32) >> 5) * 8192;
      cp16(&sK[buf ^ 1][tid * 8], kblk + tid * 8);
#if HAVE_TDM
      if (wave == 0) tdm_load_1d(&sV[buf ^ 1][0], vblk, 8192 * 2);
#else
#pragma unroll
      for (int r = 0; r < 8; ++r)
        cp16(&sV[buf ^ 1][(r * 128 + tid) * 8], vblk + (r * 128 + tid) * 8);
#endif
      if (j0 + 64 < HW)              // global_prefetch_b8 two blocks ahead
        __builtin_prefetch(Vb + ((size_t)(j0 + 64) >> 5) * 8192 + tid * 64, 0, 3);
    }

    // ---- S = Q * K^T, two 16-key subtiles (operands: one v16h each) ----
    const v16h kb0 = *(const v16h*)(&sK[buf][0]   + lane * 16);
    const v16h kb1 = *(const v16h*)(&sK[buf][512] + lane * 16);
    v8f s0 = wmma_f16(qa, kb0, vzero);
    v8f s1 = wmma_f16(qa, kb1, vzero);

    // ---- online softmax; row stats live in the 16 lanes of each group ----
    float p0[8], p1[8];
#pragma unroll
    for (int i = 0; i < 8; ++i) {
      float m = fmaxf(s0[i], s1[i]);
#pragma unroll
      for (int off = 1; off < 16; off <<= 1)
        m = fmaxf(m, __shfl_xor(m, off, 32));
      const float nm    = fmaxf(rmax[i], m);
      const float scale = __expf(rmax[i] - nm);
      p0[i] = __expf(s0[i] - nm);
      p1[i] = __expf(s1[i] - nm);
      float rs = p0[i] + p1[i];
#pragma unroll
      for (int off = 1; off < 16; off <<= 1)
        rs += __shfl_xor(rs, off, 32);
      rsum[i] = rsum[i] * scale + rs;
      rmax[i] = nm;
#pragma unroll
      for (int t = 0; t < 16; ++t) oacc[t][i] *= scale;
    }

    // ---- scatter P directly into A-operand layout (wave-private LDS) ----
#pragma unroll
    for (int i = 0; i < 8; ++i) {
      const int m = i + grp * 8;
      myP[(gc * 16 + m) * 16 + eb]     = (_Float16)p0[i];  // key = col
      myP[(gc * 16 + m) * 16 + eb + 8] = (_Float16)p1[i];  // key = col+16
    }
    const v16h pa = *(const v16h*)(myP + lane * 16);

    // ---- O += P * V over 16 channel tiles (each B operand: one v16h) ----
#pragma unroll
    for (int t = 0; t < 16; ++t) {
      const v16h vb = *(const v16h*)(&sV[buf][t * 512] + lane * 16);
      oacc[t] = wmma_f16(pa, vb, oacc[t]);
    }
  }

  // ---- epilogue: out[b,c,i] = gamma/rowsum * O + input (8 rcps, then FMA) --
  const float g = gamma[0];
  float inv[8];
#pragma unroll
  for (int i = 0; i < 8; ++i) inv[i] = g / rsum[i];
  const float* inb  = inp + (size_t)b * CC * HW;
  float*       outb = out + (size_t)b * CC * HW;
#pragma unroll
  for (int t = 0; t < 16; ++t) {
    const int c = t * 16 + col;
#pragma unroll
    for (int i = 0; i < 8; ++i) {
      const int m = i + grp * 8;
      const size_t idx = (size_t)c * HW + (size_t)(i0 + m);
      outb[idx] = fmaf(oacc[t][i], inv[i], inb[idx]);
    }
  }
}

// ---------------------------------------------------------------------------
extern "C" void kernel_launch(void* const* d_in, const int* in_sizes, int n_in,
                              void* d_out, int out_size, void* d_ws, size_t ws_size,
                              hipStream_t stream) {
  const float* inp   = (const float*)d_in[0];
  const float* orig  = (const float*)d_in[1];
  const float* wq    = (const float*)d_in[2];
  const float* bq    = (const float*)d_in[3];
  const float* wk    = (const float*)d_in[4];
  const float* bk    = (const float*)d_in[5];
  const float* wv    = (const float*)d_in[6];
  const float* bv    = (const float*)d_in[7];
  const float* gamma = (const float*)d_in[8];
  float* out = (float*)d_out;

  const int B = in_sizes[0] / (CC * HW);

  // Workspace: Q [B,HW,32] | K [B,HW,32] | V [B,HW,256], all swizzled f16.
  _Float16* qws = (_Float16*)d_ws;
  _Float16* kws = qws + (size_t)B * HW * DQK;
  _Float16* vws = kws + (size_t)B * HW * DQK;

  proj_f16_kernel<<<dim3(HW / 16, DQK / 16, B), dim3(32), 0, stream>>>(
      inp, wq, bq, qws, CC, 0);
  proj_f16_kernel<<<dim3(HW / 16, DQK / 16, B), dim3(32), 0, stream>>>(
      inp, wk, bk, kws, CC, 0);
  proj_f16_kernel<<<dim3(HW / 16, CC / 16, B), dim3(32), 0, stream>>>(
      orig, wv, bv, vws, CC, 1);

  attn_kernel<<<dim3(HW / 64, B), dim3(128), 0, stream>>>(
      qws, kws, vws, inp, gamma, out);
}
